// MultiHeadAttention_40200893890977
// MI455X (gfx1250) — matrix operand order repaired and checked by reading, not compiled, not run on hardware
//
#include <hip/hip_runtime.h>
#include <hip/hip_fp16.h>

// Problem constants (match reference)
#define DM 1024      // d_model
#define NH 16        // heads
#define DK 64        // d_k == d_v
#define BB 2         // batch
#define SS 2048      // sequence
#define LN_EPS 1e-5f

typedef __attribute__((ext_vector_type(16))) _Float16 v16h;
typedef __attribute__((ext_vector_type(8)))  float    v8f;

// ---------------------------------------------------------------------------
// CDNA5 async global->LDS copy (ASYNCcnt path, ISA 15.18.3 opcodes 95-98).
// ---------------------------------------------------------------------------
__device__ __forceinline__ void async_copy_b128(void* lds_dst, const void* gsrc) {
  unsigned ldsoff = (unsigned)(unsigned long long)lds_dst;
  unsigned long long ga = (unsigned long long)gsrc;
  asm volatile("global_load_async_to_lds_b128 %0, %1, off"
               :: "v"(ldsoff), "v"(ga) : "memory");
}
__device__ __forceinline__ void async_wait0() {
  asm volatile("s_wait_asynccnt 0x0" ::: "memory");
}
__device__ __forceinline__ void ds_wait0() {
  asm volatile("s_wait_dscnt 0x0" ::: "memory");
}

// ---------------------------------------------------------------------------
// WMMA fragment helpers (V_WMMA_F32_16X16X32_F16, wave32)
// A 16x32 (MxK): lanes 0-15 M=lane, VGPR v: K = hi*8 + 2v (+16 jump at v=4)
// B 32x16 (KxN): col = lane&15, VGPR v: K = hi*8 + 2v (+16 jump at v=4)
// C/D 16x16:     col = lane&15, VGPR r: M = r + 8*(lane>>4)
// ---------------------------------------------------------------------------
__device__ __forceinline__ int ka_map(int v, int hi) {
  return (v < 4) ? (hi * 8 + 2 * v) : (16 + hi * 8 + 2 * (v - 4));
}
__device__ __forceinline__ int kb_map(int v, int hi) { return ka_map(v, hi); }

__device__ __forceinline__ v8f wmma_f16(v16h a, v16h b, v8f c) {
  return __builtin_amdgcn_wmma_f32_16x16x32_f16(false, a, false, b, (short)0, c,
                                                false, false);
}

__device__ __forceinline__ v16h afrag_f32(const float* A, long lda,
                                          int row0, int kb, int lane) {
  int r = lane & 15, hi = lane >> 4;
  const float* p = A + (long)(row0 + r) * lda + kb;
  v16h a;
#pragma unroll
  for (int v = 0; v < 8; ++v) {
    int k = ka_map(v, hi);
    a[2 * v]     = (_Float16)p[k];
    a[2 * v + 1] = (_Float16)p[k + 1];
  }
  return a;
}

__device__ __forceinline__ v16h afrag_f16(const _Float16* A, long lda,
                                          int row0, int kb, int lane) {
  int r = lane & 15, hi = lane >> 4;
  const _Float16* p = A + (long)(row0 + r) * lda + kb;
  v16h a;
#pragma unroll
  for (int v = 0; v < 8; ++v) {
    int k = ka_map(v, hi);
    a[2 * v]     = p[k];
    a[2 * v + 1] = p[k + 1];
  }
  return a;
}

__device__ __forceinline__ v16h bfrag_f16(const _Float16* Bp, long ldb,
                                          int kb, int col0, int lane) {
  int c = col0 + (lane & 15), hi = lane >> 4;
  v16h b;
#pragma unroll
  for (int v = 0; v < 8; ++v) {
    int k = kb + kb_map(v, hi);
    b[2 * v]     = Bp[(long)k * ldb + c];
    b[2 * v + 1] = Bp[(long)(k + 1) * ldb + c];
  }
  return b;
}

// B(k,n) = M[n][k], M row-major f16 (for Q·K^T): contiguous K pairs per row.
__device__ __forceinline__ v16h bfrag_f16_T(const _Float16* M, long ldm,
                                            int kb, int col0, int lane) {
  int c = col0 + (lane & 15), hi = lane >> 4;
  const _Float16* p = M + (long)c * ldm + kb;
  v16h b;
#pragma unroll
  for (int v = 0; v < 8; ++v) {
    int k = kb_map(v, hi);
    b[2 * v]     = p[k];
    b[2 * v + 1] = p[k + 1];
  }
  return b;
}

// ---------------------------------------------------------------------------
// Kernel 0: one-time f32 -> f16 weight conversion.
// ---------------------------------------------------------------------------
__global__ __launch_bounds__(256) void f32_to_f16_kernel(const float* __restrict__ src,
                                                         _Float16* __restrict__ dst) {
  int i = (blockIdx.x * 256 + threadIdx.x) * 4;
#pragma unroll
  for (int j = 0; j < 4; ++j) dst[i + j] = (_Float16)src[i + j];
}

// ---------------------------------------------------------------------------
// Kernel 1: QKV projection.  X[B*S,1024] @ Wh(f16) + bias -> head-major f16
// Out[b][h][s][d]  (1/sqrt(64) folded for Q).  A chunk async-staged in LDS.
// ---------------------------------------------------------------------------
__global__ __launch_bounds__(128) void proj_kernel(const float* __restrict__ X,
                                                   const _Float16* __restrict__ Wh,
                                                   const float* __restrict__ bias,
                                                   _Float16* __restrict__ Outh,
                                                   float scale) {
  __shared__ __align__(16) float As[16 * 32];   // 2KB staged A chunk

  const int wave = threadIdx.x >> 5, lane = threadIdx.x & 31;
  const int t = blockIdx.x * 4 + wave;   // 4096 wave tiles
  const int mt = t >> 4;                 // uniform across the block's 4 waves
  const int nb = t & 15;                 // per-wave 64-col block
  const int m0 = mt * 16, n0 = nb * 64;
  const int cc = lane & 15, hi = lane >> 4;

  const int srow = threadIdx.x >> 3;     // 16 rows
  const int sseg = threadIdx.x & 7;      // 8 x 4 floats

  v8f c[4];
#pragma unroll
  for (int j = 0; j < 4; ++j) {
    float bv = bias[n0 + 16 * j + cc];
#pragma unroll
    for (int r = 0; r < 8; ++r) c[j][r] = bv;
  }

  for (int kb = 0; kb < DM; kb += 32) {
    __syncthreads();
    async_copy_b128(&As[srow * 32 + sseg * 4],
                    X + (long)(m0 + srow) * DM + kb + sseg * 4);
    async_wait0();
    __syncthreads();

    v16h a = afrag_f32(As, 32, 0, 0, lane);
#pragma unroll
    for (int j = 0; j < 4; ++j) {
      v16h b = bfrag_f16(Wh, DM, kb, n0 + 16 * j, lane);
      c[j] = wmma_f16(a, b, c[j]);
    }
  }

#pragma unroll
  for (int j = 0; j < 4; ++j)
#pragma unroll
    for (int r = 0; r < 8; ++r) {
      int m = m0 + r + hi * 8;           // token row in [0, B*S)
      int n = n0 + 16 * j + cc;          // output col in [0, 1024)
      int b = m >> 11, s = m & (SS - 1);
      int h = n >> 6,  d = n & 63;
      Outh[(((long)(b * NH + h) * SS + s) << 6) + d] = (_Float16)(c[j][r] * scale);
    }
}

// ---------------------------------------------------------------------------
// Kernel 2 (fully fused attention): scores -> LDS, softmax from LDS, write
// probabilities (only attn HBM traffic), context via WMMA.
// Block = one (bh, 16-query tile), 4 waves, dynamic LDS ~98KB:
//   Qs[16x64] f16 (2KB, async-staged)
//   Ls[16x2048] f16 logits (64KB)        -- never touches HBM
//   Vs[4][32x64] f16 per-wave V chunks (16KB, async-staged)
//   ctxR[4][16x64] f32 partial context (16KB)
// Wave w owns keys [w*512,(w+1)*512).  Kh/Vh are L2-resident (256KB/head
// shared by 128 blocks), so HBM attn traffic collapses to the single 536MB
// probability write.
// ---------------------------------------------------------------------------
__global__ __launch_bounds__(128) void fused_attn_kernel(
    const _Float16* __restrict__ Qh, const _Float16* __restrict__ Kh,
    const _Float16* __restrict__ Vh, const unsigned char* __restrict__ mask,
    float* __restrict__ attn, _Float16* __restrict__ Ch) {
  extern __shared__ __align__(16) char smem[];
  _Float16* Qs  = (_Float16*)smem;                          // 2048 B
  _Float16* Ls  = (_Float16*)(smem + 2048);                 // 65536 B
  _Float16* Vsb = (_Float16*)(smem + 2048 + 65536);         // 16384 B
  float*    ctxR = (float*)(smem + 2048 + 65536 + 16384);   // 16384 B
  float*    rowM = (float*)(smem + 2048 + 65536 + 16384 + 16384);  // 64 B
  float*    rowS = rowM + 16;                                      // 64 B

  const int wave = threadIdx.x >> 5, lane = threadIdx.x & 31;
  const int qt = blockIdx.x & 127;       // 128 q tiles
  const int bh = blockIdx.x >> 7;        // 32 (b,h)
  const int q0 = qt * 16;
  const int cc = lane & 15, hi = lane >> 4;
  const int bb = bh >> 4, h = bh & 15;

  const _Float16* Qp = Qh + (long)bh * SS * DK;
  const _Float16* Kp = Kh + (long)bh * SS * DK;
  const _Float16* Vp = Vh + (long)bh * SS * DK;

  // ---- stage Q tile (2KB): 128 threads x 16B ----
  {
    const int srow = threadIdx.x >> 3;   // 16 rows
    const int sseg = threadIdx.x & 7;    // 8 x 8 halves
    async_copy_b128(&Qs[srow * 64 + sseg * 8],
                    Qp + (long)(q0 + srow) * DK + sseg * 8);
    async_wait0();
  }
  __syncthreads();

  // ---- Phase 1: logits for this wave's 512-key range -> LDS ----
  const int kbeg = wave * 512;
  for (int kblk = 0; kblk < 8; ++kblk) {
    const int k0 = kbeg + kblk * 64;
    v8f sc[4];
#pragma unroll
    for (int j = 0; j < 4; ++j)
#pragma unroll
      for (int r = 0; r < 8; ++r) sc[j][r] = 0.f;
#pragma unroll
    for (int kb2 = 0; kb2 < DK; kb2 += 32) {
      v16h a = afrag_f16(Qs, 64, 0, kb2, lane);
#pragma unroll
      for (int j = 0; j < 4; ++j) {
        v16h b = bfrag_f16_T(Kp, DK, kb2, k0 + 16 * j, lane);
        sc[j] = wmma_f16(a, b, sc[j]);
      }
    }
#pragma unroll
    for (int j = 0; j < 4; ++j)
#pragma unroll
      for (int r = 0; r < 8; ++r) {
        int row = r + hi * 8;
        int k = k0 + 16 * j + cc;
        float val = sc[j][r];
        if (mask[((long)bb * SS + q0 + row) * SS + k]) val = -60000.f;  // f16-safe
        Ls[row * SS + k] = (_Float16)val;
      }
  }
  __syncthreads();

  // ---- Phase 2: row max + sum(exp) entirely from LDS (8 threads/row) ----
  {
    const int r = threadIdx.x >> 3, it = threadIdx.x & 7;
    const _Float16* lp = &Ls[r * SS + it * 256];
    float m = -3.4e38f;
    for (int i = 0; i < 256; ++i) m = fmaxf(m, (float)lp[i]);
#pragma unroll
    for (int off = 4; off > 0; off >>= 1) m = fmaxf(m, __shfl_xor(m, off, 8));
    float s = 0.f;
    for (int i = 0; i < 256; ++i) s += __expf((float)lp[i] - m);
#pragma unroll
    for (int off = 4; off > 0; off >>= 1) s += __shfl_xor(s, off, 8);
    if (it == 0) { rowM[r] = m; rowS[r] = s; }
  }
  __syncthreads();

  // ---- Phase 3: probabilities -> d_out; partial context via WMMA ----
  const float mrow = rowM[cc];           // lane's A-frag row is cc
  const float inv = 1.0f / rowS[cc];
  float* Arow = attn + ((long)bh * SS + q0 + cc) * SS;
  _Float16* Vsw = Vsb + wave * (32 * 64);  // per-wave private V slot

  v8f c[4];
#pragma unroll
  for (int j = 0; j < 4; ++j)
#pragma unroll
    for (int r = 0; r < 8; ++r) c[j][r] = 0.f;

  for (int kb = kbeg; kb < kbeg + 512; kb += 32) {
    ds_wait0();                          // prior LDS reads done before overwrite
    // stage this wave's 32x64 f16 V chunk (4KB): 32 lanes x 8 x 16B
#pragma unroll
    for (int i = 0; i < 8; ++i) {
      int idx = lane + 32 * i;           // 0..255
      int row = idx >> 3, seg = idx & 7;
      async_copy_b128(&Vsw[row * 64 + seg * 8],
                      Vp + (long)(kb + row) * DK + seg * 8);
    }
    async_wait0();

    // probability A-fragment from LDS logits; emit final p to d_out
    v16h a;
#pragma unroll
    for (int v = 0; v < 8; ++v) {
      int k = kb + ka_map(v, hi);
      float p0 = __expf((float)Ls[cc * SS + k]     - mrow) * inv;
      float p1 = __expf((float)Ls[cc * SS + k + 1] - mrow) * inv;
      Arow[k]     = p0;
      Arow[k + 1] = p1;
      a[2 * v]     = (_Float16)p0;
      a[2 * v + 1] = (_Float16)p1;
    }
#pragma unroll
    for (int j = 0; j < 4; ++j) {
      v16h b = bfrag_f16(Vsw, 64, 0, 16 * j, lane);
      c[j] = wmma_f16(a, b, c[j]);
    }
  }

  // ---- Phase 4: reduce 4 partial contexts, write Ch (token-major) ----
#pragma unroll
  for (int j = 0; j < 4; ++j)
#pragma unroll
    for (int r = 0; r < 8; ++r)
      ctxR[wave * 1024 + (r + hi * 8) * 64 + 16 * j + cc] = c[j][r];
  __syncthreads();

  for (int o = threadIdx.x; o < 1024; o += 128) {
    float s = ctxR[o] + ctxR[1024 + o] + ctxR[2048 + o] + ctxR[3072 + o];
    int r = o >> 6, d = o & 63;
    long tok = (long)bb * SS + (q0 + r);
    Ch[tok * DM + h * 64 + d] = (_Float16)s;
  }
}

// ---------------------------------------------------------------------------
// Kernel 3: preLN = Ch @ Who(f16) + bo.  A chunk async-staged into LDS.
// ---------------------------------------------------------------------------
__global__ __launch_bounds__(128) void outproj_kernel(const _Float16* __restrict__ Ch,
                                                      const _Float16* __restrict__ Who,
                                                      const float* __restrict__ bo,
                                                      float* __restrict__ preLN) {
  __shared__ __align__(16) _Float16 Cs[16 * 32];  // 1KB staged A chunk

  const int wave = threadIdx.x >> 5, lane = threadIdx.x & 31;
  const int t = blockIdx.x * 4 + wave;   // 4096 wave tiles
  const int mt = t >> 4, nb = t & 15;    // mt uniform per block
  const int m0 = mt * 16, n0 = nb * 64;
  const int cc = lane & 15, hi = lane >> 4;

  v8f c[4];
#pragma unroll
  for (int j = 0; j < 4; ++j) {
    float bv = bo[n0 + 16 * j + cc];
#pragma unroll
    for (int r = 0; r < 8; ++r) c[j][r] = bv;
  }

  for (int kb = 0; kb < DM; kb += 32) {
    __syncthreads();
    if (threadIdx.x < 64) {              // 64 threads x 16B = 1024B
      int row = threadIdx.x >> 2;        // 16 rows
      int seg = threadIdx.x & 3;         // 4 x 8 halves
      async_copy_b128(&Cs[row * 32 + seg * 8],
                      Ch + (long)(m0 + row) * DM + kb + seg * 8);
    }
    async_wait0();
    __syncthreads();

    v16h a = afrag_f16(Cs, 32, 0, 0, lane);
#pragma unroll
    for (int j = 0; j < 4; ++j) {
      v16h b = bfrag_f16(Who, DM, kb, n0 + 16 * j, lane);
      c[j] = wmma_f16(a, b, c[j]);
    }
  }

#pragma unroll
  for (int j = 0; j < 4; ++j)
#pragma unroll
    for (int r = 0; r < 8; ++r) {
      int m = m0 + r + hi * 8;
      int n = n0 + 16 * j + cc;
      preLN[(long)m * DM + n] = c[j][r];
    }
}

// ---------------------------------------------------------------------------
// Kernel 4: out = LayerNorm(preLN + Q) * gamma + beta.  One row per block.
// ---------------------------------------------------------------------------
__device__ __forceinline__ float block_sum(float val, float* sm) {
#pragma unroll
  for (int off = 16; off > 0; off >>= 1) val += __shfl_xor(val, off, 32);
  const int wave = threadIdx.x >> 5;
  if ((threadIdx.x & 31) == 0) sm[wave] = val;
  __syncthreads();
  float tot = 0.f;
#pragma unroll
  for (int i = 0; i < 8; ++i) tot += sm[i];
  __syncthreads();
  return tot;
}

__global__ __launch_bounds__(256) void ln_kernel(const float* __restrict__ preLN,
                                                 const float* __restrict__ Qin,
                                                 const float* __restrict__ gamma,
                                                 const float* __restrict__ beta,
                                                 float* __restrict__ out) {
  __shared__ float sm[8];
  const long row = blockIdx.x;           // 4096 rows
  const float* x = preLN + row * DM;
  const float* q = Qin + row * DM;

  float v[4];
  float s = 0.f;
#pragma unroll
  for (int i = 0; i < 4; ++i) {
    int col = threadIdx.x + 256 * i;
    v[i] = x[col] + q[col];
    s += v[i];
  }
  float mu = block_sum(s, sm) * (1.0f / DM);

  float va = 0.f;
#pragma unroll
  for (int i = 0; i < 4; ++i) {
    float d = v[i] - mu;
    va += d * d;
  }
  float var = block_sum(va, sm) * (1.0f / DM);
  float rstd = rsqrtf(var + LN_EPS);

#pragma unroll
  for (int i = 0; i < 4; ++i) {
    int col = threadIdx.x + 256 * i;
    out[row * DM + col] = (v[i] - mu) * rstd * gamma[col] + beta[col];
  }
}

// ---------------------------------------------------------------------------
// Host launcher
// ---------------------------------------------------------------------------
extern "C" void kernel_launch(void* const* d_in, const int* in_sizes, int n_in,
                              void* d_out, int out_size, void* d_ws, size_t ws_size,
                              hipStream_t stream) {
  const float* Q     = (const float*)d_in[0];
  const float* K     = (const float*)d_in[1];
  const float* V     = (const float*)d_in[2];
  const unsigned char* mask = (const unsigned char*)d_in[3];
  const float* Wq    = (const float*)d_in[4];
  const float* bq    = (const float*)d_in[5];
  const float* Wk    = (const float*)d_in[6];
  const float* bk    = (const float*)d_in[7];
  const float* Wv    = (const float*)d_in[8];
  const float* bv    = (const float*)d_in[9];
  const float* Wo    = (const float*)d_in[10];
  const float* bo    = (const float*)d_in[11];
  const float* gamma = (const float*)d_in[12];
  const float* beta  = (const float*)d_in[13];

  float* out  = (float*)d_out;
  float* attn = out + (long)BB * SS * DM;   // attn starts after output tensor

  const long tokElems = (long)BB * SS * DM; // 4,194,304
  const long wElems   = (long)DM * DM;      // 1,048,576
  _Float16* Qh  = (_Float16*)d_ws;          // head-major [B,H,S,64]
  _Float16* Kh  = Qh + tokElems;
  _Float16* Vh  = Kh + tokElems;
  _Float16* Ch  = Vh + tokElems;            // token-major [B*S,1024]
  float* preLN  = (float*)(Ch + tokElems);  // [B*S,1024]
  _Float16* Whq = (_Float16*)(preLN + tokElems);
  _Float16* Whk = Whq + wElems;
  _Float16* Whv = Whk + wElems;
  _Float16* Who = Whv + wElems;

  const float qscale = 0.125f;              // 1/sqrt(D_K)

  f32_to_f16_kernel<<<1024, 256, 0, stream>>>(Wq, Whq);
  f32_to_f16_kernel<<<1024, 256, 0, stream>>>(Wk, Whk);
  f32_to_f16_kernel<<<1024, 256, 0, stream>>>(Wv, Whv);
  f32_to_f16_kernel<<<1024, 256, 0, stream>>>(Wo, Who);

  proj_kernel<<<1024, 128, 0, stream>>>(Q, Whq, bq, Qh, qscale);
  proj_kernel<<<1024, 128, 0, stream>>>(K, Whk, bk, Kh, 1.0f);
  proj_kernel<<<1024, 128, 0, stream>>>(V, Whv, bv, Vh, 1.0f);

  // dynamic LDS: Qs(2048) + Ls(65536) + Vs(16384) + ctxR(16384) + rowM/S(128)
  const size_t smemBytes = 2048 + 65536 + 16384 + 16384 + 128;
  fused_attn_kernel<<<4096, 128, smemBytes, stream>>>(Qh, Kh, Vh, mask, attn, Ch);

  outproj_kernel<<<1024, 128, 0, stream>>>(Ch, Who, bo, preLN);
  ln_kernel<<<4096, 256, 0, stream>>>(preLN, Q, gamma, beta, out);
}
